// TrajectoryGenerator_18614388260983
// MI455X (gfx1250) — matrix operand order repaired and checked by reading, not compiled, not run on hardware
//
#include <hip/hip_runtime.h>

// ---------------------------------------------------------------------------
// TrajectoryGenerator on gfx1250: batch-tiled LSTM enc/dec with bf16 WMMA.
// Each wave handles M=16 batch rows; gate GEMMs (16x256, K=32/64) are done
// with v_wmma_f32_16x16x32_bf16 against weights resident in LDS (bf16).
// ---------------------------------------------------------------------------

typedef __attribute__((ext_vector_type(16))) __bf16 v16bf;
typedef __attribute__((ext_vector_type(8)))  __bf16 v8bf;
typedef __attribute__((ext_vector_type(8)))  float  v8f;

#define DEVINL __device__ __forceinline__

// ---- model dims -----------------------------------------------------------
constexpr int Bsz      = 65536;
constexpr int OBS_LEN  = 20;
constexpr int PRED_LEN = 30;
constexpr int E        = 32;
constexpr int H        = 64;
constexpr int FOURH    = 256;
constexpr int COND_PAD = 160;   // 144 padded to 5*32

// ---- LDS layout (bytes) ---------------------------------------------------
constexpr size_t OFF_WIH_E = 0;                                  // 256x32 bf16
constexpr size_t OFF_WHH_E = OFF_WIH_E + FOURH * E * 2;          // 256x64 bf16
constexpr size_t OFF_WIH_D = OFF_WHH_E + FOURH * H * 2;
constexpr size_t OFF_WHH_D = OFF_WIH_D + FOURH * E * 2;
constexpr size_t OFF_WH    = OFF_WHH_D + FOURH * H * 2;          // 64x160 bf16
constexpr size_t OFF_WC    = OFF_WH + H * COND_PAD * 2;
constexpr size_t OFF_STG   = OFF_WC + H * COND_PAD * 2;          // 8 waves x 16x160 bf16
constexpr size_t OFF_BENC  = OFF_STG + 8 * 16 * COND_PAD * 2;
constexpr size_t OFF_BDEC  = OFF_BENC + FOURH * 4;
constexpr size_t OFF_BH    = OFF_BDEC + FOURH * 4;
constexpr size_t OFF_BC    = OFF_BH + H * 4;
constexpr size_t OFF_WEMB  = OFF_BC + H * 4;                     // 32x2 f32
constexpr size_t OFF_BEMB  = OFF_WEMB + 64 * 4;
constexpr size_t OFF_WOUT  = OFF_BEMB + 32 * 4;                  // 2x64 f32
constexpr size_t OFF_BOUT  = OFF_WOUT + 128 * 4;
constexpr size_t SMEM_TOTAL = OFF_BOUT + 16;

// ---- helpers --------------------------------------------------------------
DEVINL v8f wmma_bf16(v16bf a, v16bf b, v8f c) {
  return __builtin_amdgcn_wmma_f32_16x16x32_bf16(false, a, false, b,
                                                 (short)0, c, false, false);
}

DEVINL v16bf load_frag2(const __bf16* p0, const __bf16* p1) {
  union { v16bf v; v8bf h[2]; } f;
  f.h[0] = *reinterpret_cast<const v8bf*>(p0);
  f.h[1] = *reinterpret_cast<const v8bf*>(p1);
  return f.v;
}

// A fragment (16x32) of row-major S[16][stride] starting at column kb.
// Lane layout (ISA 7.12.2, 16-bit A): elems 0..7 = K kb+8*half+{0..7},
// elems 8..15 = K kb+16+8*half+{0..7}, row m = lane%16.
DEVINL v16bf load_a_frag(const __bf16* S, int stride, int m, int kb, int half) {
  const __bf16* p0 = S + m * stride + kb + 8 * half;
  return load_frag2(p0, p0 + 16);
}

// B fragment (32x16) from LDS weight W[N][Ks] (K contiguous): lane n holds
// K kb+8*half+{0..7} then kb+16+8*half+{0..7} (kb16 = kb + 16*half).
DEVINL v16bf load_b_frag(const __bf16* W, int Ks, int ncol, int kb16) {
  const __bf16* p = W + ncol * Ks + (kb16 & ~31) + 8 * ((kb16 >> 4) & 1);
  return load_frag2(p, p + 16);
}

DEVINL float sigm(float x)      { return 1.0f / (1.0f + __expf(-x)); }
DEVINL float tanh_fast(float x) { return 2.0f / (1.0f + __expf(-2.0f * x)) - 1.0f; }

// acc[16] covers N=0..255: acc += A(16x32) x B(32x16 per tile)
DEVINL void gemm_all_nt(v8f (&acc)[16], v16bf a, const __bf16* W, int Ks,
                        int kb, int lane16, int half) {
  const int kb16 = kb + 16 * half;
  #pragma unroll
  for (int nt = 0; nt < 16; ++nt)
    acc[nt] = wmma_bf16(a, load_b_frag(W, Ks, nt * 16 + lane16, kb16), acc[nt]);
}

// Embedding A fragment computed directly in A layout: emb[m][e] =
// relu(x0*W[e][0] + x1*W[e][1] + b[e]) for this lane's 16 K(=e) slots.
DEVINL v16bf emb_frag(float x0, float x1, const float* sWemb,
                      const float* sBemb, int half) {
  v16bf a;
  #pragma unroll
  for (int i = 0; i < 8; ++i) {
    const int e0 = 8 * half + i;
    const int e1 = 16 + 8 * half + i;
    float v0 = fmaxf(fmaf(x1, sWemb[e0 * 2 + 1], fmaf(x0, sWemb[e0 * 2], sBemb[e0])), 0.0f);
    float v1 = fmaxf(fmaf(x1, sWemb[e1 * 2 + 1], fmaf(x0, sWemb[e1 * 2], sBemb[e1])), 0.0f);
    a[i]     = (__bf16)v0;
    a[8 + i] = (__bf16)v1;
  }
  return a;
}

// PyTorch gate order i,f,g,o along N: tiles [0..3]=i [4..7]=f [8..11]=g [12..15]=o
DEVINL void lstm_update(const v8f (&acc)[16], v8f (&h)[4], v8f (&c)[4]) {
  #pragma unroll
  for (int j = 0; j < 4; ++j) {
    #pragma unroll
    for (int r = 0; r < 8; ++r) {
      float iv = sigm(acc[j][r]);
      float fv = sigm(acc[4 + j][r]);
      float gv = tanh_fast(acc[8 + j][r]);
      float ov = sigm(acc[12 + j][r]);
      float cn = fmaf(fv, c[j][r], iv * gv);
      c[j][r] = cn;
      h[j][r] = ov * tanh_fast(cn);
    }
  }
}

// Scatter h (C-layout tiles) into per-wave stage as bf16 rows [m][n].
DEVINL void store_h(const v8f (&h)[4], __bf16* stg, int lane16, int half) {
  #pragma unroll
  for (int j = 0; j < 4; ++j) {
    #pragma unroll
    for (int r = 0; r < 8; ++r)
      stg[(r + 8 * half) * COND_PAD + j * 16 + lane16] = (__bf16)h[j][r];
  }
}

// ---------------------------------------------------------------------------
__global__ __launch_bounds__(256) void traj_gen_kernel(
    const float* __restrict__ obs, const float* __restrict__ noise,
    const float* __restrict__ scene,
    const float* __restrict__ W_emb, const float* __restrict__ b_emb,
    const float* __restrict__ Wih_enc, const float* __restrict__ Whh_enc,
    const float* __restrict__ b_enc,
    const float* __restrict__ W_h, const float* __restrict__ b_h,
    const float* __restrict__ W_c, const float* __restrict__ b_c,
    const float* __restrict__ Wih_dec, const float* __restrict__ Whh_dec,
    const float* __restrict__ b_dec,
    const float* __restrict__ W_out, const float* __restrict__ b_out,
    float* __restrict__ out) {
  extern __shared__ char smem[];
  __bf16* sWihE = (__bf16*)(smem + OFF_WIH_E);
  __bf16* sWhhE = (__bf16*)(smem + OFF_WHH_E);
  __bf16* sWihD = (__bf16*)(smem + OFF_WIH_D);
  __bf16* sWhhD = (__bf16*)(smem + OFF_WHH_D);
  __bf16* sWh   = (__bf16*)(smem + OFF_WH);
  __bf16* sWc   = (__bf16*)(smem + OFF_WC);
  __bf16* sStg  = (__bf16*)(smem + OFF_STG);
  float*  sBenc = (float*)(smem + OFF_BENC);
  float*  sBdec = (float*)(smem + OFF_BDEC);
  float*  sBh   = (float*)(smem + OFF_BH);
  float*  sBc   = (float*)(smem + OFF_BC);
  float*  sWemb = (float*)(smem + OFF_WEMB);
  float*  sBemb = (float*)(smem + OFF_BEMB);
  float*  sWout = (float*)(smem + OFF_WOUT);
  float*  sBout = (float*)(smem + OFF_BOUT);

  const int tid = threadIdx.x;
  const int wave = tid >> 5, lane = tid & 31;
  const int lane16 = lane & 15, half = lane >> 4;

  // ---- stage weights into LDS (f32 -> bf16) ----
  for (int i = tid; i < FOURH * E; i += 256) sWihE[i] = (__bf16)Wih_enc[i];
  for (int i = tid; i < FOURH * H; i += 256) sWhhE[i] = (__bf16)Whh_enc[i];
  for (int i = tid; i < FOURH * E; i += 256) sWihD[i] = (__bf16)Wih_dec[i];
  for (int i = tid; i < FOURH * H; i += 256) sWhhD[i] = (__bf16)Whh_dec[i];
  for (int i = tid; i < H * COND_PAD; i += 256) {
    int n = i / COND_PAD, k = i % COND_PAD;
    sWh[i] = (k < 144) ? (__bf16)W_h[n * 144 + k] : (__bf16)0.0f;
    sWc[i] = (k < 144) ? (__bf16)W_c[n * 144 + k] : (__bf16)0.0f;
  }
  if (tid < FOURH) { sBenc[tid] = b_enc[tid]; sBdec[tid] = b_dec[tid]; }
  if (tid < H)     { sBh[tid] = b_h[tid]; sBc[tid] = b_c[tid]; }
  if (tid < 64)    sWemb[tid] = W_emb[tid];
  if (tid < 32)    sBemb[tid] = b_emb[tid];
  if (tid < 128)   sWout[tid] = W_out[tid];
  if (tid < 2)     sBout[tid] = b_out[tid];
  __syncthreads();

  const int gm = blockIdx.x * 128 + wave * 16 + lane16;  // this lane's A-row
  __bf16* stg = sStg + wave * (16 * COND_PAD);           // per-wave staging

  // init stage: h0 = 0 in cols [0,64); cond tail = [noise | scene | 0-pad]
  if (half == 0) {
    for (int k = 0; k < 64; ++k)  stg[lane16 * COND_PAD + k] = (__bf16)0.0f;
    for (int k = 0; k < 16; ++k)  stg[lane16 * COND_PAD + 64 + k]  = (__bf16)noise[gm * 16 + k];
    for (int k = 0; k < 64; ++k)  stg[lane16 * COND_PAD + 80 + k]  = (__bf16)scene[gm * 64 + k];
    for (int k = 0; k < 16; ++k)  stg[lane16 * COND_PAD + 144 + k] = (__bf16)0.0f;
  }

  v8f h[4], c[4];
  #pragma unroll
  for (int j = 0; j < 4; ++j) {
    #pragma unroll
    for (int r = 0; r < 8; ++r) { h[j][r] = 0.0f; c[j][r] = 0.0f; }
  }

  // ---------------- encoder: 20 steps ----------------
  #pragma unroll 1
  for (int t = 0; t < OBS_LEN; ++t) {
    __syncthreads();
    v8f acc[16];
    #pragma unroll
    for (int nt = 0; nt < 16; ++nt) {
      float bv = sBenc[nt * 16 + lane16];
      #pragma unroll
      for (int r = 0; r < 8; ++r) acc[nt][r] = bv;
    }
    float x0 = obs[gm * (OBS_LEN * 2) + t * 2 + 0];
    float x1 = obs[gm * (OBS_LEN * 2) + t * 2 + 1];
    v16bf ax = emb_frag(x0, x1, sWemb, sBemb, half);
    gemm_all_nt(acc, ax, sWihE, E, 0, lane16, half);
    v16bf ah0 = load_a_frag(stg, COND_PAD, lane16, 0, half);
    v16bf ah1 = load_a_frag(stg, COND_PAD, lane16, 32, half);
    gemm_all_nt(acc, ah0, sWhhE, H, 0, lane16, half);
    gemm_all_nt(acc, ah1, sWhhE, H, 32, lane16, half);
    lstm_update(acc, h, c);
    __syncthreads();
    store_h(h, stg, lane16, half);
  }
  __syncthreads();

  // ---------------- cond -> decoder initial (h, c): (16x160)@(160x64) ------
  {
    v8f hacc[4], cacc[4];
    #pragma unroll
    for (int j = 0; j < 4; ++j) {
      float bh = sBh[j * 16 + lane16], bc = sBc[j * 16 + lane16];
      #pragma unroll
      for (int r = 0; r < 8; ++r) { hacc[j][r] = bh; cacc[j][r] = bc; }
    }
    #pragma unroll
    for (int kc = 0; kc < 5; ++kc) {
      v16bf a = load_a_frag(stg, COND_PAD, lane16, kc * 32, half);
      const int kb16 = kc * 32 + 16 * half;
      #pragma unroll
      for (int j = 0; j < 4; ++j) {
        hacc[j] = wmma_bf16(a, load_b_frag(sWh, COND_PAD, j * 16 + lane16, kb16), hacc[j]);
        cacc[j] = wmma_bf16(a, load_b_frag(sWc, COND_PAD, j * 16 + lane16, kb16), cacc[j]);
      }
    }
    #pragma unroll
    for (int j = 0; j < 4; ++j) { h[j] = hacc[j]; c[j] = cacc[j]; }
  }
  __syncthreads();
  store_h(h, stg, lane16, half);

  // ---------------- decoder: 30 autoregressive steps ----------------
  float x0 = obs[gm * (OBS_LEN * 2) + (OBS_LEN - 1) * 2 + 0];
  float x1 = obs[gm * (OBS_LEN * 2) + (OBS_LEN - 1) * 2 + 1];
  #pragma unroll 1
  for (int t = 0; t < PRED_LEN; ++t) {
    __syncthreads();
    v8f acc[16];
    #pragma unroll
    for (int nt = 0; nt < 16; ++nt) {
      float bv = sBdec[nt * 16 + lane16];
      #pragma unroll
      for (int r = 0; r < 8; ++r) acc[nt][r] = bv;
    }
    v16bf ax = emb_frag(x0, x1, sWemb, sBemb, half);
    gemm_all_nt(acc, ax, sWihD, E, 0, lane16, half);
    v16bf ah0 = load_a_frag(stg, COND_PAD, lane16, 0, half);
    v16bf ah1 = load_a_frag(stg, COND_PAD, lane16, 32, half);
    gemm_all_nt(acc, ah0, sWhhD, H, 0, lane16, half);
    gemm_all_nt(acc, ah1, sWhhD, H, 32, lane16, half);
    lstm_update(acc, h, c);
    __syncthreads();
    store_h(h, stg, lane16, half);
    __syncthreads();

    // output head y = h @ W_out.T + b_out (row m read back from stage)
    float y0 = sBout[0], y1 = sBout[1];
    #pragma unroll
    for (int n = 0; n < 64; ++n) {
      float hv = (float)stg[lane16 * COND_PAD + n];
      y0 = fmaf(hv, sWout[n], y0);
      y1 = fmaf(hv, sWout[64 + n], y1);
    }
    if (half == 0) {
      float2 yv; yv.x = y0; yv.y = y1;
      reinterpret_cast<float2*>(out)[gm * PRED_LEN + t] = yv;
    }
    x0 = y0; x1 = y1;  // autoregressive feedback
  }
}

// ---------------------------------------------------------------------------
extern "C" void kernel_launch(void* const* d_in, const int* in_sizes, int n_in,
                              void* d_out, int out_size, void* d_ws, size_t ws_size,
                              hipStream_t stream) {
  (void)in_sizes; (void)n_in; (void)out_size; (void)d_ws; (void)ws_size;
  const float* obs     = (const float*)d_in[0];
  const float* noise   = (const float*)d_in[1];
  const float* scene   = (const float*)d_in[2];
  const float* W_emb   = (const float*)d_in[3];
  const float* b_emb   = (const float*)d_in[4];
  const float* Wih_enc = (const float*)d_in[5];
  const float* Whh_enc = (const float*)d_in[6];
  const float* b_enc   = (const float*)d_in[7];
  const float* W_h     = (const float*)d_in[8];
  const float* b_h     = (const float*)d_in[9];
  const float* W_c     = (const float*)d_in[10];
  const float* b_c     = (const float*)d_in[11];
  const float* Wih_dec = (const float*)d_in[12];
  const float* Whh_dec = (const float*)d_in[13];
  const float* b_dec   = (const float*)d_in[14];
  const float* W_out   = (const float*)d_in[15];
  const float* b_out   = (const float*)d_in[16];
  float* out = (float*)d_out;

  dim3 grid(Bsz / 128), block(256);
  traj_gen_kernel<<<grid, block, SMEM_TOTAL, stream>>>(
      obs, noise, scene, W_emb, b_emb, Wih_enc, Whh_enc, b_enc,
      W_h, b_h, W_c, b_c, Wih_dec, Whh_dec, b_dec, W_out, b_out, out);
}
